// COCLE_13554916786253
// MI455X (gfx1250) — matrix operand order repaired and checked by reading, not compiled, not run on hardware
//
#include <hip/hip_runtime.h>
#include <hip/hip_bf16.h>
#include <cstdint>
#include <cstddef>

// ---------------- problem constants ----------------
#define NN    50000
#define HD    128
#define NE    800000
#define NM    10000
#define NNZV  800000
#define NPOSV 64
#define TAU_INV 2.0f     // 1/0.5
#define ALPHA_C 1.0f
#define LAM_C   0.5f
#define EPS_C   1e-8f

typedef __bf16 bf16x16 __attribute__((ext_vector_type(16)));
typedef float  f32x8   __attribute__((ext_vector_type(8)));
typedef unsigned short u16x8 __attribute__((ext_vector_type(8)));

__device__ __forceinline__ unsigned short f2bf(float f) {
  union { float f; unsigned u; } c; c.f = f;
  unsigned u = c.u;
  unsigned r = u + 0x7FFFu + ((u >> 16) & 1u);   // round-to-nearest-even
  return (unsigned short)(r >> 16);
}

__device__ __forceinline__ void atomAddF(float* p, float v) {
  (void)__hip_atomic_fetch_add(p, v, __ATOMIC_RELAXED, __HIP_MEMORY_SCOPE_AGENT);
}

__device__ __forceinline__ float warpSum(float v) {
#pragma unroll
  for (int o = 16; o > 0; o >>= 1) v += __shfl_xor(v, o, 32);
  return v;
}

// ---------------- elementwise / setup kernels ----------------
__global__ void k_fill(float* __restrict__ p, float v, int n) {
  int i = blockIdx.x * blockDim.x + threadIdx.x;
  if (i < n) p[i] = v;
}

__global__ void k_count(float* __restrict__ cnt, const int* __restrict__ idx, int n) {
  int i = blockIdx.x * blockDim.x + threadIdx.x;
  if (i < n) atomAddF(&cnt[idx[i]], 1.0f);
}

__global__ void k_rsqrt_inplace(float* p, int n) {
  int i = blockIdx.x * blockDim.x + threadIdx.x;
  if (i < n) p[i] = (p[i] > 0.f) ? rsqrtf(p[i]) : 0.f;
}

__global__ void k_recip_inplace(float* p, int n) {
  int i = blockIdx.x * blockDim.x + threadIdx.x;
  if (i < n) p[i] = (p[i] > 0.f) ? (1.0f / p[i]) : 0.f;
}

__global__ void k_edge_norm(float* __restrict__ nrm, const int* __restrict__ src,
                            const int* __restrict__ dst, const float* __restrict__ dinv, int n) {
  int i = blockIdx.x * blockDim.x + threadIdx.x;
  if (i < n) nrm[i] = dinv[src[i]] * dinv[dst[i]];
}

// xw for one-hot query input: out[i,:] = (i==q) ? Wrow : 0
__global__ void k_onehot(float* __restrict__ out, const float* __restrict__ row,
                         const int* __restrict__ qp, int nh) {
  int i = blockIdx.x * blockDim.x + threadIdx.x;
  if (i < nh) out[i] = ((i >> 7) == qp[0]) ? row[i & 127] : 0.f;
}

// ---------------- WMMA bf16 GEMM: C[N,128] = A[N,128] @ W[128,128] ----------------
// flags bit0: add bias[col]   bit1: relu
__global__ __launch_bounds__(256) void k_gemm128(const float* __restrict__ A,
                                                 const float* __restrict__ W,
                                                 const float* __restrict__ bias,
                                                 float* __restrict__ C,
                                                 int nrows, int flags) {
  __shared__ unsigned short As[128 * 136];   // A tile, bf16, padded stride 136
  __shared__ unsigned short Wt[128 * 136];   // W transposed: Wt[c][k], bf16
  const int t = threadIdx.x;

  // stage W (row-major [k][c]) transposed into LDS as bf16
  for (int i = t; i < 128 * 128; i += 256) {
    int k = i >> 7, c = i & 127;
    Wt[c * 136 + k] = f2bf(W[i]);
  }
  // stage A tile (128 rows), 2 threads/row, 64 cols each, f32 -> bf16
  const int row0 = blockIdx.x * 128;
  {
    int r = t >> 1;
    int cbase = (t & 1) * 64;
    int gr = row0 + r;
    if (gr < nrows) {
      const float4* src = (const float4*)(A + (size_t)gr * 128 + cbase);
#pragma unroll
      for (int j = 0; j < 16; ++j) {
        float4 v = src[j];
        int c = cbase + j * 4;
        As[r * 136 + c + 0] = f2bf(v.x);
        As[r * 136 + c + 1] = f2bf(v.y);
        As[r * 136 + c + 2] = f2bf(v.z);
        As[r * 136 + c + 3] = f2bf(v.w);
      }
    } else {
#pragma unroll
      for (int j = 0; j < 64; ++j) As[r * 136 + cbase + j] = 0;
    }
  }
  __syncthreads();

  const int wave = t >> 5, lane = t & 31;
  const int lrow   = (wave << 4) + (lane & 15);  // A row within tile for this lane
  const int kbaseA = (lane >> 4) << 3;           // 0 or 8  (ISA 16-bit A 16x32 layout)
  const int kbaseB = (lane >> 4) << 3;           // 0 or 8  (same k order as A)

  f32x8 acc[8] = {};
  union AB { bf16x16 v; u16x8 h[2]; };

#pragma unroll
  for (int ks = 0; ks < 4; ++ks) {               // K = 128 in 4 steps of 32
    const int kofs = ks * 32;
    AB a;
    a.h[0] = *(const u16x8*)&As[lrow * 136 + kofs + kbaseA];
    a.h[1] = *(const u16x8*)&As[lrow * 136 + kofs + kbaseA + 16];
#pragma unroll
    for (int ct = 0; ct < 8; ++ct) {             // 8 column tiles of 16
      const int c = (ct << 4) + (lane & 15);
      AB b;
      b.h[0] = *(const u16x8*)&Wt[c * 136 + kofs + kbaseB];
      b.h[1] = *(const u16x8*)&Wt[c * 136 + kofs + kbaseB + 16];
      acc[ct] = __builtin_amdgcn_wmma_f32_16x16x32_bf16(
          false, a.v, false, b.v, (short)0, acc[ct], false, false);
    }
  }

  // C/D layout: VGPR v -> row (v + 8*(lane>=16)), col = ct*16 + (lane&15)
  const int rbase = row0 + (wave << 4) + ((lane >> 4) << 3);
  const int cN = lane & 15;
#pragma unroll
  for (int ct = 0; ct < 8; ++ct) {
    const int col = (ct << 4) + cN;
    const float bv = (flags & 1) ? bias[col] : 0.f;
#pragma unroll
    for (int v = 0; v < 8; ++v) {
      int gr = rbase + v;
      if (gr < nrows) {
        float x = acc[ct][v] + bv;
        if (flags & 2) x = fmaxf(x, 0.f);
        C[(size_t)gr * 128 + col] = x;
      }
    }
  }
}

// ---------------- scatter kernels (one wave per edge, 4 feats/lane) ----------------
__global__ __launch_bounds__(256) void k_gcn_scatter(float* __restrict__ acc,
    const float* __restrict__ xw, const int* __restrict__ src,
    const int* __restrict__ dst, const float* __restrict__ nrm, int n) {
  int wid = blockIdx.x * 8 + (threadIdx.x >> 5);
  if (wid >= n) return;
  int lane = threadIdx.x & 31;
  int s = src[wid], d = dst[wid];
  float w = nrm[wid];
  const float* xs = xw + (size_t)s * HD;
  float* ad = acc + (size_t)d * HD;
#pragma unroll
  for (int r = 0; r < 4; ++r) { int f = lane + (r << 5); atomAddF(&ad[f], xs[f] * w); }
}

__global__ __launch_bounds__(256) void k_hscat1(float* __restrict__ me,
    const float* __restrict__ xw, const int* __restrict__ ni,
    const int* __restrict__ ei, int n) {
  int wid = blockIdx.x * 8 + (threadIdx.x >> 5);
  if (wid >= n) return;
  int lane = threadIdx.x & 31;
  int s = ni[wid], d = ei[wid];
  const float* xs = xw + (size_t)s * HD;
  float* ad = me + (size_t)d * HD;
#pragma unroll
  for (int r = 0; r < 4; ++r) { int f = lane + (r << 5); atomAddF(&ad[f], xs[f]); }
}

__global__ __launch_bounds__(256) void k_hscat2(float* __restrict__ acc,
    const float* __restrict__ me, const int* __restrict__ ni,
    const int* __restrict__ ei, const float* __restrict__ binv, int n) {
  int wid = blockIdx.x * 8 + (threadIdx.x >> 5);
  if (wid >= n) return;
  int lane = threadIdx.x & 31;
  int nd = ni[wid], e = ei[wid];
  float w = binv[e];
  const float* ms = me + (size_t)e * HD;
  float* ad = acc + (size_t)nd * HD;
#pragma unroll
  for (int r = 0; r < 4; ++r) { int f = lane + (r << 5); atomAddF(&ad[f], ms[f] * w); }
}

// ---------------- finish kernels ----------------
__global__ void k_gcn_finish(float* __restrict__ out, const float* __restrict__ acc,
    const float* __restrict__ xw, const float* __restrict__ dinv,
    const float* __restrict__ bias, const float* __restrict__ res, int relu, int nh) {
  int i = blockIdx.x * blockDim.x + threadIdx.x;
  if (i >= nh) return;
  int node = i >> 7;
  float di = dinv[node];
  float v = acc[i] + xw[i] * di * di + bias[i & 127];   // self-loop term + bias
  if (res) v += res[i];
  if (relu) v = fmaxf(v, 0.f);
  out[i] = v;
}

__global__ void k_h_finish(float* __restrict__ out, const float* __restrict__ acc,
    const float* __restrict__ Dinv, const float* __restrict__ bias,
    const float* __restrict__ res, int relu, int nh) {
  int i = blockIdx.x * blockDim.x + threadIdx.x;
  if (i >= nh) return;
  int node = i >> 7;
  float v = acc[i] * Dinv[node] + bias[i & 127];
  if (res) v += res[i];
  if (relu) v = fmaxf(v, 0.f);
  out[i] = v;
}

// ---------------- fuse (one wave per node) ----------------
__global__ __launch_bounds__(256) void k_fuse(float* __restrict__ out,
    const float* __restrict__ xa, const float* __restrict__ xb,
    const float* __restrict__ wa, const float* __restrict__ wb, int n) {
  int wid = blockIdx.x * 8 + (threadIdx.x >> 5);
  if (wid >= n) return;
  int lane = threadIdx.x & 31;
  const float* pa = xa + (size_t)wid * HD;
  const float* pb = xb + (size_t)wid * HD;
  float va[4], vb[4], sa = 0.f, sb = 0.f;
#pragma unroll
  for (int r = 0; r < 4; ++r) {
    int f = lane + (r << 5);
    va[r] = pa[f]; vb[r] = pb[f];
    sa += va[r] * wa[f]; sb += vb[r] * wb[f];
  }
  sa = warpSum(sa); sb = warpSum(sb);
  float m = fmaxf(sa, sb);
  float ea = expf(sa - m), eb = expf(sb - m);
  float inv = 1.f / (ea + eb);
  float* po = out + (size_t)wid * HD;
#pragma unroll
  for (int r = 0; r < 4; ++r) {
    int f = lane + (r << 5);
    po[f] = ea * inv * va[r] + eb * inv * vb[r];
  }
}

// fuse with virtual qlin:  xa[i,:] = lqb + (i==q)*lqW
__global__ __launch_bounds__(256) void k_fuse_q(float* __restrict__ out,
    const float* __restrict__ xb, const float* __restrict__ lqW,
    const float* __restrict__ lqb, const int* __restrict__ qp,
    const float* __restrict__ wa, const float* __restrict__ wb, int n) {
  int wid = blockIdx.x * 8 + (threadIdx.x >> 5);
  if (wid >= n) return;
  int lane = threadIdx.x & 31;
  int isq = (wid == qp[0]);
  const float* pb = xb + (size_t)wid * HD;
  float va[4], vb[4], sa = 0.f, sb = 0.f;
#pragma unroll
  for (int r = 0; r < 4; ++r) {
    int f = lane + (r << 5);
    va[r] = lqb[f] + (isq ? lqW[f] : 0.f);
    vb[r] = pb[f];
    sa += va[r] * wa[f]; sb += vb[r] * wb[f];
  }
  sa = warpSum(sa); sb = warpSum(sb);
  float m = fmaxf(sa, sb);
  float ea = expf(sa - m), eb = expf(sb - m);
  float inv = 1.f / (ea + eb);
  float* po = out + (size_t)wid * HD;
#pragma unroll
  for (int r = 0; r < 4; ++r) {
    int f = lane + (r << 5);
    po[f] = ea * inv * va[r] + eb * inv * vb[r];
  }
}

// ---------------- loss kernels ----------------
__global__ __launch_bounds__(256) void k_rownorm(float* __restrict__ rn,
                                                 const float* __restrict__ x, int n) {
  int wid = blockIdx.x * 8 + (threadIdx.x >> 5);
  if (wid >= n) return;
  int lane = threadIdx.x & 31;
  const float* p = x + (size_t)wid * HD;
  float s = 0.f;
#pragma unroll
  for (int r = 0; r < 4; ++r) { float v = p[lane + (r << 5)]; s += v * v; }
  s = warpSum(s);
  if (lane == 0) rn[wid] = fmaxf(sqrtf(s), EPS_C);
}

// s4[0..N)=s_ii  [N..2N)=s_aa  [2N..3N)=s_ia  [3N..4N)=s_ai
__global__ __launch_bounds__(256) void k_logits(float* __restrict__ s4,
    const float* __restrict__ hf, const float* __restrict__ haf,
    const float* __restrict__ rnh, const float* __restrict__ rna,
    const int* __restrict__ qp, int n) {
  int wid = blockIdx.x * 8 + (threadIdx.x >> 5);
  if (wid >= n) return;
  int lane = threadIdx.x & 31;
  int q = qp[0];
  const float* hq = hf  + (size_t)q * HD;
  const float* aq = haf + (size_t)q * HD;
  const float* hi = hf  + (size_t)wid * HD;
  const float* ai = haf + (size_t)wid * HD;
  float d0 = 0, d1 = 0, d2 = 0, d3 = 0;
#pragma unroll
  for (int r = 0; r < 4; ++r) {
    int f = lane + (r << 5);
    float hqv = hq[f], aqv = aq[f], hiv = hi[f], aiv = ai[f];
    d0 += hqv * hiv; d1 += aqv * aiv; d2 += hqv * aiv; d3 += aqv * hiv;
  }
  d0 = warpSum(d0); d1 = warpSum(d1); d2 = warpSum(d2); d3 = warpSum(d3);
  if (lane == 0) {
    float rq = rnh[q], raq = rna[q], ri = rnh[wid], rai = rna[wid];
    s4[0 * NN + wid] = d0 / (rq  * ri ) * TAU_INV;
    s4[1 * NN + wid] = d1 / (raq * rai) * TAU_INV;
    s4[2 * NN + wid] = d2 / (rq  * rai) * TAU_INV;
    s4[3 * NN + wid] = d3 / (raq * ri ) * TAU_INV;
  }
}

__global__ __launch_bounds__(1024) void k_loss(float* __restrict__ out,
    const float* __restrict__ s4, const int* __restrict__ pos,
    const int* __restrict__ qp, int n) {
  __shared__ float sbuf[1024];
  __shared__ float gmax[4], lse[4];
  __shared__ int posP[NPOSV], posV[NPOSV];
  const int t = threadIdx.x;
  const int q = qp[0];

  // pass 1: max of each logit array
  float m[4] = {-3.4e38f, -3.4e38f, -3.4e38f, -3.4e38f};
  for (int i = t; i < n; i += 1024)
#pragma unroll
    for (int a = 0; a < 4; ++a) m[a] = fmaxf(m[a], s4[a * NN + i]);
  for (int a = 0; a < 4; ++a) {
    sbuf[t] = m[a]; __syncthreads();
    for (int o = 512; o > 0; o >>= 1) {
      if (t < o) sbuf[t] = fmaxf(sbuf[t], sbuf[t + o]);
      __syncthreads();
    }
    if (t == 0) gmax[a] = sbuf[0];
    __syncthreads();
  }
  // pass 2: sum exp(s - max)
  float s[4] = {0, 0, 0, 0};
  for (int i = t; i < n; i += 1024)
#pragma unroll
    for (int a = 0; a < 4; ++a) s[a] += expf(s4[a * NN + i] - gmax[a]);
  for (int a = 0; a < 4; ++a) {
    sbuf[t] = s[a]; __syncthreads();
    for (int o = 512; o > 0; o >>= 1) {
      if (t < o) sbuf[t] += sbuf[t + o];
      __syncthreads();
    }
    if (t == 0) lse[a] = gmax[a] + logf(sbuf[0]);
    __syncthreads();
  }
  // positives mask: dedupe pos, exclude q (mirrors .at[pos].set(True).at[q].set(False))
  if (t < NPOSV) {
    int p = pos[t];
    int dup = 0;
    for (int k = 0; k < t; ++k) if (pos[k] == p) dup = 1;
    posP[t] = p;
    posV[t] = (!dup && p != q) ? 1 : 0;
  }
  __syncthreads();
  if (t == 0) {
    float npos = 0.f, sp[4] = {0, 0, 0, 0};
    for (int j = 0; j < NPOSV; ++j) if (posV[j]) {
      npos += 1.f;
      for (int a = 0; a < 4; ++a) sp[a] += s4[a * NN + posP[j]];
    }
    npos = fmaxf(npos, 1.f);
    float intra = 0.5f * ((lse[0] - sp[0] / npos) + (lse[1] - sp[1] / npos));
    float inter = 0.5f * ((lse[2] - sp[2] / npos) + (lse[3] - sp[3] / npos));
    float unsup = 0.5f * (lse[2] - s4[2 * NN + q]) + 0.5f * (lse[3] - s4[3 * NN + q]);
    out[0] = intra + ALPHA_C * inter + LAM_C * unsup;
  }
}

// ---------------- host orchestration ----------------
extern "C" void kernel_launch(void* const* d_in, const int* in_sizes, int n_in,
                              void* d_out, int out_size, void* d_ws, size_t ws_size,
                              hipStream_t stream) {
  (void)n_in; (void)out_size;
  // --- detect flattening convention ---
  const float* feats; const int *eidx, *eaug, *qp, *pos;
  const float* prm[40];
  int pbase;
  if (in_sizes[0] == 2 * NE) {                // fully sorted top-level keys
    eidx = (const int*)d_in[0];
    eaug = (const int*)d_in[1];
    feats = (const float*)d_in[2];
    pbase = 3;
    pos = (const int*)d_in[43];
    qp  = (const int*)d_in[44];
  } else {                                    // setup_inputs() dict order
    feats = (const float*)d_in[0];
    pbase = 1;
    eidx = (const int*)d_in[41];
    eaug = (const int*)d_in[42];
    qp   = (const int*)d_in[43];
    pos  = (const int*)d_in[44];
  }
  for (int i = 0; i < 40; ++i) prm[i] = (const float*)d_in[pbase + i];
  const bool sortedP = (in_sizes[pbase] == HD * HD);  // W0 (16384) first iff sorted keys
  const float *W0,*W1,*aW0,*aW1,*aatt0,*aatt1,*ab0,*ab1,*afW0,*afW1,*afatt,*afb0,*afb1,
      *afqatt,*aqW0,*aqW1,*aqatt0,*aqatt1,*aqb0,*aqb1,*att0,*att1,*b0,*b1,*fW0,*fW1,
      *fatt,*fb0,*fb1,*fqatt,*lfW,*lfb,*lqW,*lqb,*qW0,*qW1,*qatt0,*qatt1,*qb0,*qb1;
  if (sortedP) {
    W0=prm[0];W1=prm[1];aW0=prm[2];aW1=prm[3];aatt0=prm[4];aatt1=prm[5];ab0=prm[6];
    ab1=prm[7];afW0=prm[8];afW1=prm[9];afatt=prm[10];afb0=prm[11];afb1=prm[12];
    afqatt=prm[13];aqW0=prm[14];aqW1=prm[15];aqatt0=prm[16];aqatt1=prm[17];aqb0=prm[18];
    aqb1=prm[19];att0=prm[20];att1=prm[21];b0=prm[22];b1=prm[23];fW0=prm[24];fW1=prm[25];
    fatt=prm[26];fb0=prm[27];fb1=prm[28];fqatt=prm[29];lfW=prm[30];lfb=prm[31];
    lqW=prm[32];lqb=prm[33];qW0=prm[34];qW1=prm[35];qatt0=prm[36];qatt1=prm[37];
    qb0=prm[38];qb1=prm[39];
  } else {                                    // insertion order
    qW0=prm[0];qb0=prm[1];qW1=prm[2];qb1=prm[3];W0=prm[4];b0=prm[5];W1=prm[6];b1=prm[7];
    fW0=prm[8];fb0=prm[9];fW1=prm[10];fb1=prm[11];aqW0=prm[12];aqb0=prm[13];aqW1=prm[14];
    aqb1=prm[15];aW0=prm[16];ab0=prm[17];aW1=prm[18];ab1=prm[19];afW0=prm[20];
    afb0=prm[21];afW1=prm[22];afb1=prm[23];lqW=prm[24];lqb=prm[25];lfW=prm[26];
    lfb=prm[27];qatt0=prm[28];qatt1=prm[29];att0=prm[30];att1=prm[31];fqatt=prm[32];
    fatt=prm[33];aqatt0=prm[34];aqatt1=prm[35];aatt0=prm[36];aatt1=prm[37];
    afqatt=prm[38];afatt=prm[39];
  }

  // --- workspace carve-out ---
  float* w = (float*)d_ws;
  size_t off = 0;
  auto alloc = [&](size_t nfl) { float* p = w + off; off += (nfl + 3) & ~(size_t)3; return p; };
  const size_t NHf = (size_t)NN * HD;
  float* dinvG = alloc(NN);
  float* DinvH = alloc(NN);
  float* Binv  = alloc(NM);
  float* enorm = alloc(NE);
  float* XW  = alloc(NHf);
  float* ACC = alloc(NHf);
  float* ME  = alloc((size_t)NM * HD);
  float* hq  = alloc(NHf);
  float* h_  = alloc(NHf);
  float* hf  = alloc(NHf);
  float* haq = alloc(NHf);
  float* ha  = alloc(NHf);
  float* haf = alloc(NHf);
  float* flin= alloc(NHf);
  float* rnh = alloc(NN);
  float* rna = alloc(NN);
  float* s4  = alloc((size_t)4 * NN);
  if (ws_size < off * sizeof(float)) return;   // cannot run (deterministic no-op)

  const int NH = NN * HD, MH = NM * HD;
  dim3 b256(256);
  auto blks = [](int n, int per) { return dim3((unsigned)((n + per - 1) / per)); };
  auto fill = [&](float* p, float v, int n) {
    k_fill<<<blks(n, 256), b256, 0, stream>>>(p, v, n);
  };
  auto gemm = [&](float* out, const float* A, const float* Wm, const float* bias, int flags) {
    k_gemm128<<<blks(NN, 128), b256, 0, stream>>>(A, Wm, bias, out, NN, flags);
  };
  auto gcn = [&](float* out, const float* bias, const float* res, int relu) {
    fill(ACC, 0.f, NH);
    k_gcn_scatter<<<blks(NE, 8), b256, 0, stream>>>(ACC, XW, eidx, eidx + NE, enorm, NE);
    k_gcn_finish<<<blks(NH, 256), b256, 0, stream>>>(out, ACC, XW, dinvG, bias, res, relu, NH);
  };
  auto hconv = [&](float* out, const float* bias, const float* res, int relu) {
    fill(ME, 0.f, MH);
    k_hscat1<<<blks(NNZV, 8), b256, 0, stream>>>(ME, XW, eaug, eaug + NNZV, NNZV);
    fill(ACC, 0.f, NH);
    k_hscat2<<<blks(NNZV, 8), b256, 0, stream>>>(ACC, ME, eaug, eaug + NNZV, Binv, NNZV);
    k_h_finish<<<blks(NH, 256), b256, 0, stream>>>(out, ACC, DinvH, bias, res, relu, NH);
  };
  auto fuse = [&](float* out, const float* xa, const float* xb, const float* wa, const float* wb) {
    k_fuse<<<blks(NN, 8), b256, 0, stream>>>(out, xa, xb, wa, wb, NN);
  };

  // --- graph / hypergraph normalizations ---
  fill(dinvG, 1.f, NN);                                      // self loops
  k_count<<<blks(NE, 256), b256, 0, stream>>>(dinvG, eidx + NE, NE);
  k_rsqrt_inplace<<<blks(NN, 256), b256, 0, stream>>>(dinvG, NN);
  k_edge_norm<<<blks(NE, 256), b256, 0, stream>>>(enorm, eidx, eidx + NE, dinvG, NE);
  fill(DinvH, 0.f, NN);
  k_count<<<blks(NNZV, 256), b256, 0, stream>>>(DinvH, eaug, NNZV);
  k_recip_inplace<<<blks(NN, 256), b256, 0, stream>>>(DinvH, NN);
  fill(Binv, 0.f, NM);
  k_count<<<blks(NNZV, 256), b256, 0, stream>>>(Binv, eaug + NNZV, NNZV);
  k_recip_inplace<<<blks(NM, 256), b256, 0, stream>>>(Binv, NM);

  // ---------------- layer 0 ----------------
  k_onehot<<<blks(NH, 256), b256, 0, stream>>>(XW, qW0, qp, NH);
  gcn(hq, qb0, nullptr, 1);
  gemm(XW, feats, W0, nullptr, 0);
  gcn(h_, b0, nullptr, 1);
  fuse(hf, hq, h_, qatt0, att0);
  k_onehot<<<blks(NH, 256), b256, 0, stream>>>(XW, aqW0, qp, NH);
  hconv(haq, aqb0, nullptr, 1);
  gemm(XW, feats, aW0, nullptr, 0);
  hconv(ha, ab0, nullptr, 1);
  fuse(haf, haq, ha, aqatt0, aatt0);
  gemm(flin, feats, lfW, lfb, /*bias*/1);
  k_fuse_q<<<blks(NN, 8), b256, 0, stream>>>(ACC, flin, lqW, lqb, qp, fqatt, fatt, NN);  // hf_
  gemm(XW, ACC, fW0, nullptr, 0);
  gcn(hf, fb0, hf, 1);                                       // hf = relu(hf + gcn(hf_))
  k_fuse_q<<<blks(NN, 8), b256, 0, stream>>>(ACC, flin, lqW, lqb, qp, afqatt, afatt, NN); // hfh_
  gemm(XW, ACC, afW0, nullptr, 0);
  hconv(haf, afb0, haf, 1);                                  // haf = relu(haf + hconv(hfh_))

  // ---------------- layer 1 ----------------
  gemm(XW, hq, qW1, nullptr, 0); gcn(hq, qb1, nullptr, 1);
  gemm(XW, h_, W1,  nullptr, 0); gcn(h_, b1, nullptr, 1);
  fuse(hq, hq, h_, qatt1, att1);                             // hf1 -> hq (in place ok)
  gemm(XW, haq, aqW1, nullptr, 0); hconv(haq, aqb1, nullptr, 1);
  gemm(XW, ha,  aW1,  nullptr, 0); hconv(ha,  ab1,  nullptr, 1);
  fuse(haq, haq, ha, aqatt1, aatt1);                         // haf1 -> haq
  gemm(XW, hf, fW1, nullptr, 0);
  gcn(hf, fb1, hq, 1);                                       // hf = relu(hf1 + gcn(hf))
  gemm(XW, haf, afW1, nullptr, 0);
  hconv(haf, afb1, haq, 1);                                  // haf = relu(haf1 + hconv(haf))

  // ---------------- contrastive loss ----------------
  k_rownorm<<<blks(NN, 8), b256, 0, stream>>>(rnh, hf, NN);
  k_rownorm<<<blks(NN, 8), b256, 0, stream>>>(rna, haf, NN);
  k_logits<<<blks(NN, 8), b256, 0, stream>>>(s4, hf, haf, rnh, rna, qp, NN);
  k_loss<<<1, 1024, 0, stream>>>((float*)d_out, s4, pos, qp, NN);
}
